// PolicyWithValue_63900523430200
// MI455X (gfx1250) — matrix operand order repaired and checked by reading, not compiled, not run on hardware
//
#include <hip/hip_runtime.h>
#include <hip/hip_bf16.h>

// ---------------- model constants (match reference) ----------------
#define BB   2
#define SS   512
#define DDIM 1024
#define HH   16
#define FF   4096
#define VV   32000
#define LLAY 2
#define RR   16
#define DHH  64           // DDIM / HH
#define LORA_SCALE 2.0f   // LORA_ALPHA / R = 32/16

typedef __attribute__((ext_vector_type(16))) _Float16 v16h;
typedef __attribute__((ext_vector_type(8)))  float    v8f;
typedef __attribute__((ext_vector_type(4)))  float    v4f;
typedef __attribute__((ext_vector_type(2)))  float    v2f;
typedef __attribute__((ext_vector_type(4)))  int      v4i;

// ---- CDNA5 async global->LDS copy (ASYNCcnt), feature-gated ----
#if defined(__has_builtin)
#  if __has_builtin(__builtin_amdgcn_global_load_async_to_lds_b128) && \
      __has_builtin(__builtin_amdgcn_s_wait_asynccnt)
#    define USE_ASYNC_LDS 1
#  endif
#endif
#ifndef USE_ASYNC_LDS
#  define USE_ASYNC_LDS 0
#endif

#if USE_ASYNC_LDS
// builtin prototype (from hipcc diagnostic): (v4i AS1*, v4i AS3*, imm offset, imm cpol)
__device__ __forceinline__ void async_b128(const float* g, void* l)
{
    __builtin_amdgcn_global_load_async_to_lds_b128(
        (__attribute__((address_space(1))) v4i*)(g),
        (__attribute__((address_space(3))) v4i*)(l), 0, 0);
}
#endif

// =====================================================================
// FULL-TILE GEMM (hot path):  C = alpha * (A x B) [+ Res]
// Requires M%128==0, N%64==0, K%32==0 (true for every hot GEMM here).
//   A: fp32 row-major [M,K]; B: fp32 [K,N] (transB=0) or [N,K] (transB=1)
// Block tile 128x64, BK=32, 256 threads = 8 wave32 waves.
// Wave tile 32x32 -> 4 x v_wmma_f32_16x16x32_f16 per K step.
// f32 tiles staged into LDS (async b128 when available); f32->f16
// conversion happens at fragment build (co-executes with WMMA).
// =====================================================================
__global__ __launch_bounds__(256) void gemm_f16_wmma_full(
    const float* __restrict__ A, int lda, long sA1, long sA2,
    const float* __restrict__ Bm, int ldb, long sB1, long sB2, int transB,
    float* __restrict__ C, int ldc, long sC1, long sC2,
    const float* __restrict__ Res, int ldres, long sR1, long sR2,
    int M, int N, int K, int zdiv, float alpha)
{
    __shared__ float As32[128][36];   // row stride 144B (16B multiple)
    __shared__ float Bs32[32][68];    // row stride 272B (16B multiple)

    const int z = blockIdx.z;
    const long offA = (long)(z / zdiv) * sA1 + (long)(z % zdiv) * sA2;
    const long offB = (long)(z / zdiv) * sB1 + (long)(z % zdiv) * sB2;
    const long offC = (long)(z / zdiv) * sC1 + (long)(z % zdiv) * sC2;
    const long offR = (long)(z / zdiv) * sR1 + (long)(z % zdiv) * sR2;

    const int tid  = threadIdx.x;
    const int lane = tid & 31;
    const int wid  = tid >> 5;
    const int wm   = wid & 3;     // 0..3 -> 32-row quarter of 128-row tile
    const int wn   = wid >> 2;    // 0..1 -> 32-col half of 64-col tile
    const int bm0  = blockIdx.y * 128;
    const int bn0  = blockIdx.x * 64;

    v8f acc00 = {}, acc01 = {}, acc10 = {}, acc11 = {};

    // ISA 16-bit A 16x32 layout: lanes0-15 hold K 0-7 & 16-23, lanes16-31 K 8-15 & 24-31
    const int kb_a = (lane & 16) ? 8 : 0;
    // A and B fragments here both use contiguous k: lanes0-15 K 0-15, lanes16-31 K 16-31
    const int kb_b = (lane & 16) ? 16 : 0;
    const int l15  = lane & 15;

    for (int kt = 0; kt < K; kt += 32) {
        // ---- stage A tile 128x32 f32 (4 b128 chunks per thread) ----
        #pragma unroll
        for (int i = 0; i < 4; ++i) {
            int cidx = tid + i * 256;           // 0..1023
            int r    = cidx >> 3;               // 0..127
            int col  = (cidx & 7) << 2;         // 0..28
            const float* src = A + offA + (long)(bm0 + r) * lda + (kt + col);
#if USE_ASYNC_LDS
            async_b128(src, (void*)&As32[r][col]);
#else
            *(v4f*)&As32[r][col] = *(const v4f*)src;
#endif
        }
        // ---- stage B tile 32x64 f32 ----
        if (!transB) {
            #pragma unroll
            for (int i = 0; i < 2; ++i) {
                int cidx = tid + i * 256;       // 0..511
                int r    = cidx >> 4;           // 0..31
                int col  = (cidx & 15) << 2;    // 0..60
                const float* src = Bm + offB + (long)(kt + r) * ldb + (bn0 + col);
#if USE_ASYNC_LDS
                async_b128(src, (void*)&Bs32[r][col]);
#else
                *(v4f*)&Bs32[r][col] = *(const v4f*)src;
#endif
            }
        } else {
            // B stored [N,K]: contiguous in k -> vector load, scatter rows
            const int nloc = tid >> 2;          // 0..63
            const int c    = (tid & 3) << 3;    // k offset 0,8,16,24
            const float* src = Bm + offB + (long)(bn0 + nloc) * ldb + (kt + c);
            v4f f0 = *(const v4f*)src;
            v4f f1 = *(const v4f*)(src + 4);
            #pragma unroll
            for (int j = 0; j < 4; ++j) Bs32[c + j][nloc]     = f0[j];
            #pragma unroll
            for (int j = 0; j < 4; ++j) Bs32[c + 4 + j][nloc] = f1[j];
        }
#if USE_ASYNC_LDS
        __builtin_amdgcn_s_wait_asynccnt(0);
#endif
        __syncthreads();

        // prefetch next K tile into cache (global_prefetch_b8)
        if (kt + 32 < K) {
            const float* pa = A + offA + (long)(bm0 + (tid >> 1)) * lda
                              + (kt + 32 + ((tid & 1) << 4));
            __builtin_prefetch(pa, 0, 1);
            if (!transB) {
                const float* pb = Bm + offB + (long)(kt + 32 + (tid >> 3)) * ldb
                                  + (bn0 + ((tid & 7) << 3));
                __builtin_prefetch(pb, 0, 1);
            }
        }

        // ---- build f16 fragments from f32 LDS tiles ----
        v16h b0, b1;
        {
            int n0 = wn * 32 + l15;
            #pragma unroll
            for (int j = 0; j < 8; ++j) {
                int k0 = kb_b + 2 * j;
                b0[2 * j]     = (_Float16)Bs32[k0][n0];
                b0[2 * j + 1] = (_Float16)Bs32[k0 + 1][n0];
                b1[2 * j]     = (_Float16)Bs32[k0][n0 + 16];
                b1[2 * j + 1] = (_Float16)Bs32[k0 + 1][n0 + 16];
            }
        }
        v16h a0, a1;
        {
            int m0 = wm * 32 + l15;
            int m1 = m0 + 16;
            #pragma unroll
            for (int j = 0; j < 8; ++j) {
                int k0 = kb_b + 2 * j; // even
                v2f p0 = *(const v2f*)&As32[m0][k0];
                v2f p1 = *(const v2f*)&As32[m1][k0];
                a0[2 * j]     = (_Float16)p0[0];
                a0[2 * j + 1] = (_Float16)p0[1];
                a1[2 * j]     = (_Float16)p1[0];
                a1[2 * j + 1] = (_Float16)p1[1];
            }
        }
        acc00 = __builtin_amdgcn_wmma_f32_16x16x32_f16(false, a0, false, b0, (short)0, acc00, false, false);
        acc01 = __builtin_amdgcn_wmma_f32_16x16x32_f16(false, a0, false, b1, (short)0, acc01, false, false);
        acc10 = __builtin_amdgcn_wmma_f32_16x16x32_f16(false, a1, false, b0, (short)0, acc10, false, false);
        acc11 = __builtin_amdgcn_wmma_f32_16x16x32_f16(false, a1, false, b1, (short)0, acc11, false, false);
        __syncthreads();
    }

    // ---- epilogue (no bounds checks: everything tiles exactly) ----
    const int nb = bn0 + wn * 32 + l15;
    const int mb = bm0 + wm * 32 + ((lane >> 4) << 3);
    #pragma unroll
    for (int r = 0; r < 8; ++r) {
        int m0 = mb + r;
        int m1 = m0 + 16;
        float x00 = alpha * acc00[r];
        float x01 = alpha * acc01[r];
        float x10 = alpha * acc10[r];
        float x11 = alpha * acc11[r];
        if (Res) {
            x00 += Res[offR + (long)m0 * ldres + nb];
            x01 += Res[offR + (long)m0 * ldres + nb + 16];
            x10 += Res[offR + (long)m1 * ldres + nb];
            x11 += Res[offR + (long)m1 * ldres + nb + 16];
        }
        C[offC + (long)m0 * ldc + nb]      = x00;
        C[offC + (long)m0 * ldc + nb + 16] = x01;
        C[offC + (long)m1 * ldc + nb]      = x10;
        C[offC + (long)m1 * ldc + nb + 16] = x11;
    }
}

// =====================================================================
// EDGE GEMM (cold path, LoRA shapes: N=16 or K=16). 64x64 tile,
// guarded scalar staging, f16 LDS, 2 WMMAs per K step.
// =====================================================================
__global__ __launch_bounds__(256) void gemm_f16_wmma_part(
    const float* __restrict__ A, int lda,
    const float* __restrict__ Bm, int ldb, int transB,
    float* __restrict__ C, int ldc,
    const float* __restrict__ Res, int ldres,
    int M, int N, int K, float alpha)
{
    __shared__ _Float16 As[64][40];
    __shared__ _Float16 Bs[32][72];

    const int tid  = threadIdx.x;
    const int lane = tid & 31;
    const int wid  = tid >> 5;
    const int wm   = wid & 1;
    const int wn   = wid >> 1;
    const int bm0  = blockIdx.y * 64;
    const int bn0  = blockIdx.x * 64;

    v8f acc0 = {}, acc1 = {};

    const int kb_a = (lane & 16) ? 8 : 0;
    const int kb_b = (lane & 16) ? 16 : 0;
    const int l15  = lane & 15;

    for (int kt = 0; kt < K; kt += 32) {
        for (int idx = tid; idx < 64 * 32; idx += 256) {
            int r = idx >> 5, c = idx & 31;
            int gm = bm0 + r, gk = kt + c;
            float val = (gm < M && gk < K) ? A[(long)gm * lda + gk] : 0.0f;
            As[r][c] = (_Float16)val;
        }
        for (int idx = tid; idx < 32 * 64; idx += 256) {
            int r = idx >> 6, c = idx & 63;
            int gk = kt + r, gn = bn0 + c;
            float val = 0.0f;
            if (gk < K && gn < N)
                val = transB ? Bm[(long)gn * ldb + gk] : Bm[(long)gk * ldb + gn];
            Bs[r][c] = (_Float16)val;
        }
        __syncthreads();

        v16h bfrag;
        {
            int n = wn * 16 + l15;
            #pragma unroll
            for (int j = 0; j < 8; ++j) {
                int k0 = kb_b + 2 * j;
                bfrag[2 * j]     = Bs[k0][n];
                bfrag[2 * j + 1] = Bs[k0 + 1][n];
            }
        }
        v16h afrag0, afrag1;
        {
            int m0 = wm * 32 + l15;
            int m1 = m0 + 16;
            #pragma unroll
            for (int j = 0; j < 8; ++j) {
                int k0 = kb_b + 2 * j;
                afrag0[2 * j]     = As[m0][k0];
                afrag0[2 * j + 1] = As[m0][k0 + 1];
                afrag1[2 * j]     = As[m1][k0];
                afrag1[2 * j + 1] = As[m1][k0 + 1];
            }
        }
        acc0 = __builtin_amdgcn_wmma_f32_16x16x32_f16(false, afrag0, false, bfrag, (short)0, acc0, false, false);
        acc1 = __builtin_amdgcn_wmma_f32_16x16x32_f16(false, afrag1, false, bfrag, (short)0, acc1, false, false);
        __syncthreads();
    }

    int n = bn0 + wn * 16 + l15;
    int mbase = bm0 + wm * 32 + ((lane >> 4) << 3);
    if (n < N) {
        #pragma unroll
        for (int r = 0; r < 8; ++r) {
            int m = mbase + r;
            if (m < M) {
                float v0 = alpha * acc0[r];
                if (Res) v0 += Res[(long)m * ldres + n];
                C[(long)m * ldc + n] = v0;
            }
            int m2 = m + 16;
            if (m2 < M) {
                float v1 = alpha * acc1[r];
                if (Res) v1 += Res[(long)m2 * ldres + n];
                C[(long)m2 * ldc + n] = v1;
            }
        }
    }
}

// =====================================================================
// Elementwise / reduction kernels
// =====================================================================
__global__ void embed_gather_kernel(const int* __restrict__ ids,
                                    const float* __restrict__ embed,
                                    float* __restrict__ h, int n)
{
    int i = blockIdx.x * blockDim.x + threadIdx.x;
    if (i >= n) return;
    int tok = i / DDIM;
    int d   = i - tok * DDIM;
    h[i] = embed[(long)ids[tok] * DDIM + d];
}

__global__ __launch_bounds__(256) void rmsnorm_kernel(
    const float* __restrict__ x, const float* __restrict__ w,
    float* __restrict__ y, int ncols)
{
    __shared__ float red[256];
    long row = blockIdx.x;
    const float* xr = x + row * ncols;
    float s = 0.0f;
    for (int c = threadIdx.x; c < ncols; c += 256) { float v = xr[c]; s += v * v; }
    red[threadIdx.x] = s;
    __syncthreads();
    for (int st = 128; st > 0; st >>= 1) {
        if (threadIdx.x < st) red[threadIdx.x] += red[threadIdx.x + st];
        __syncthreads();
    }
    float inv = rsqrtf(red[0] / (float)ncols + 1e-5f);
    for (int c = threadIdx.x; c < ncols; c += 256)
        y[row * ncols + c] = xr[c] * inv * w[c];
}

// RoPE applied in place to q and k, both stored [B,S,H,DH] == [B,S,D]
__global__ void rope_kernel(float* __restrict__ q, float* __restrict__ k, int n)
{
    int i = blockIdx.x * blockDim.x + threadIdx.x;
    if (i >= n) return;               // n = B*S*H*(DH/2)
    const int half = DHH / 2;
    int hi   = i & (half - 1);
    int rest = i / half;
    int hd   = rest & (HH - 1);
    int tok  = rest / HH;             // b*S + s
    int s    = tok & (SS - 1);
    float inv = __expf(-((float)hi / (float)half) * 9.210340372f); // ln(10000)
    float ang = (float)s * inv;
    float c  = __cosf(ang);
    float sn = __sinf(ang);
    long base = (long)tok * DDIM + hd * DHH;
    float q1 = q[base + hi], q2 = q[base + hi + half];
    q[base + hi]        = q1 * c - q2 * sn;
    q[base + hi + half] = q2 * c + q1 * sn;
    float k1 = k[base + hi], k2 = k[base + hi + half];
    k[base + hi]        = k1 * c - k2 * sn;
    k[base + hi + half] = k2 * c + k1 * sn;
}

// Masked causal softmax over each row of scores [B*H, S, S] (in place).
__global__ __launch_bounds__(256) void softmax_kernel(
    float* __restrict__ sc, const int* __restrict__ amask)
{
    __shared__ float red[256];
    int r  = blockIdx.x;              // 0 .. B*H*S-1
    int qi = r & (SS - 1);
    int zz = r / SS;                  // b*H + h
    int b  = zz / HH;
    float* row = sc + ((long)zz * SS + qi) * SS;

    float mx = -1e30f;
    for (int j = threadIdx.x; j < SS; j += 256) {
        bool keep = (j <= qi) && (amask[b * SS + j] > 0);
        float v = row[j] + (keep ? 0.0f : -1e9f);
        row[j] = v;
        mx = fmaxf(mx, v);
    }
    red[threadIdx.x] = mx;
    __syncthreads();
    for (int st = 128; st > 0; st >>= 1) {
        if (threadIdx.x < st) red[threadIdx.x] = fmaxf(red[threadIdx.x], red[threadIdx.x + st]);
        __syncthreads();
    }
    mx = red[0];
    __syncthreads();

    float sum = 0.0f;
    for (int j = threadIdx.x; j < SS; j += 256) {
        float e = __expf(row[j] - mx);
        row[j] = e;
        sum += e;
    }
    red[threadIdx.x] = sum;
    __syncthreads();
    for (int st = 128; st > 0; st >>= 1) {
        if (threadIdx.x < st) red[threadIdx.x] += red[threadIdx.x + st];
        __syncthreads();
    }
    float invs = 1.0f / red[0];
    for (int j = threadIdx.x; j < SS; j += 256) row[j] *= invs;
}

__global__ void silu_mul_kernel(float* __restrict__ g, const float* __restrict__ u, long n)
{
    long i = (long)blockIdx.x * blockDim.x + threadIdx.x;
    if (i >= n) return;
    float x = g[i];
    g[i] = (x / (1.0f + __expf(-x))) * u[i];
}

__global__ __launch_bounds__(256) void value_head_kernel(
    const float* __restrict__ hn, const float* __restrict__ vw,
    const float* __restrict__ vb, float* __restrict__ out)
{
    __shared__ float red[256];
    long row = blockIdx.x;
    float s = 0.0f;
    for (int c = threadIdx.x; c < DDIM; c += 256) s += hn[row * DDIM + c] * vw[c];
    red[threadIdx.x] = s;
    __syncthreads();
    for (int st = 128; st > 0; st >>= 1) {
        if (threadIdx.x < st) red[threadIdx.x] += red[threadIdx.x + st];
        __syncthreads();
    }
    if (threadIdx.x == 0) out[row] = red[0] + vb[0];
}

// =====================================================================
// Host-side launch helpers
// =====================================================================
static void gemm_batched(hipStream_t st,
                         const float* A, int lda, long sA1, long sA2,
                         const float* Bm, int ldb, long sB1, long sB2, int transB,
                         float* C, int ldc, long sC1, long sC2,
                         const float* Res, int ldres, long sR1, long sR2,
                         int M, int N, int K, int batches, int zdiv, float alpha)
{
    if ((M % 128 == 0) && (N % 64 == 0) && (K % 32 == 0)) {
        dim3 grid(N / 64, M / 128, batches);
        gemm_f16_wmma_full<<<grid, 256, 0, st>>>(A, lda, sA1, sA2,
                                                 Bm, ldb, sB1, sB2, transB,
                                                 C, ldc, sC1, sC2,
                                                 Res, ldres, sR1, sR2,
                                                 M, N, K, zdiv, alpha);
    } else {
        // edge shapes are only ever non-batched here
        dim3 grid((N + 63) / 64, (M + 63) / 64, 1);
        gemm_f16_wmma_part<<<grid, 256, 0, st>>>(A, lda, Bm, ldb, transB,
                                                 C, ldc, Res, ldres,
                                                 M, N, K, alpha);
    }
}

static void gemm1(hipStream_t st, const float* A, int lda,
                  const float* Bm, int ldb, int transB,
                  float* C, int ldc, const float* Res,
                  int M, int N, int K, float alpha)
{
    gemm_batched(st, A, lda, 0, 0, Bm, ldb, 0, 0, transB,
                 C, ldc, 0, 0, Res, ldc, 0, 0, M, N, K, 1, 1, alpha);
}

// Y = X@W + (X@Aw)@Bw * LORA_SCALE [+ Res]
static void lin_lora(hipStream_t st, const float* X, const float* W,
                     const float* Aw, const float* Bw,
                     float* t1, float* Y, const float* Res,
                     int M, int Kin, int Nout)
{
    gemm1(st, X, Kin, Aw, RR, 0, t1, RR, nullptr, M, RR, Kin, 1.0f);           // t1 = X@Aw
    gemm1(st, X, Kin, W, Nout, 0, Y, Nout, Res, M, Nout, Kin, 1.0f);           // Y  = X@W (+Res)
    gemm1(st, t1, RR, Bw, Nout, 0, Y, Nout, Y, M, Nout, RR, LORA_SCALE);       // Y += s*t1@Bw
}

extern "C" void kernel_launch(void* const* d_in, const int* in_sizes, int n_in,
                              void* d_out, int out_size, void* d_ws, size_t ws_size,
                              hipStream_t stream)
{
    (void)in_sizes; (void)n_in; (void)out_size; (void)ws_size;

    const int*   ids   = (const int*)d_in[0];
    const int*   amask = (const int*)d_in[1];
    const float* embed = (const float*)d_in[2];
    const float* ln1   = (const float*)d_in[3];
    const float* ln2   = (const float*)d_in[4];
    const float* wq = (const float*)d_in[5];  const float* aq = (const float*)d_in[6];  const float* bq = (const float*)d_in[7];
    const float* wk = (const float*)d_in[8];  const float* ak = (const float*)d_in[9];  const float* bk = (const float*)d_in[10];
    const float* wv = (const float*)d_in[11]; const float* av = (const float*)d_in[12]; const float* bv = (const float*)d_in[13];
    const float* wo = (const float*)d_in[14]; const float* ao = (const float*)d_in[15]; const float* bo = (const float*)d_in[16];
    const float* wg = (const float*)d_in[17]; const float* ag = (const float*)d_in[18]; const float* bg = (const float*)d_in[19];
    const float* wu = (const float*)d_in[20]; const float* au = (const float*)d_in[21]; const float* bu = (const float*)d_in[22];
    const float* wd = (const float*)d_in[23]; const float* ad = (const float*)d_in[24]; const float* bd = (const float*)d_in[25];
    const float* final_ln  = (const float*)d_in[26];
    const float* lm_head_w = (const float*)d_in[27];
    const float* value_w   = (const float*)d_in[28];
    const float* value_b   = (const float*)d_in[29];

    float* out_logits = (float*)d_out;
    float* out_values = out_logits + (size_t)BB * SS * VV;

    // ---- workspace layout (floats) ----
    float* ws = (float*)d_ws;
    const size_t T = (size_t)BB * SS;        // 1024 tokens
    float* h  = ws;                          // [T, D]
    float* xn = h  + T * DDIM;               // [T, D]
    float* q  = xn + T * DDIM;               // [T, D]
    float* k  = q  + T * DDIM;               // [T, D]
    float* v  = k  + T * DDIM;               // [T, D]
    float* o  = v  + T * DDIM;               // [T, D]
    float* t1 = o  + T * DDIM;               // [T, R] (padded slot)
    float* sc = t1 + T * 64;                 // [B*H, S, S]
    float* g  = sc + (size_t)BB * HH * SS * SS;  // [T, F]
    float* u  = g  + T * FF;                 // [T, F]

    // ---- embedding gather ----
    {
        int n = (int)(T * DDIM);
        embed_gather_kernel<<<(n + 255) / 256, 256, 0, stream>>>(ids, embed, h, n);
    }

    for (int l = 0; l < LLAY; ++l) {
        const float* Wq = wq + (size_t)l * DDIM * DDIM;
        const float* Aq = aq + (size_t)l * DDIM * RR;
        const float* Bq = bq + (size_t)l * RR * DDIM;
        const float* Wk = wk + (size_t)l * DDIM * DDIM;
        const float* Ak = ak + (size_t)l * DDIM * RR;
        const float* Bk = bk + (size_t)l * RR * DDIM;
        const float* Wv = wv + (size_t)l * DDIM * DDIM;
        const float* Av = av + (size_t)l * DDIM * RR;
        const float* Bv = bv + (size_t)l * RR * DDIM;
        const float* Wo = wo + (size_t)l * DDIM * DDIM;
        const float* Ao = ao + (size_t)l * DDIM * RR;
        const float* Bo = bo + (size_t)l * RR * DDIM;
        const float* Wg = wg + (size_t)l * DDIM * FF;
        const float* Ag = ag + (size_t)l * DDIM * RR;
        const float* Bg = bg + (size_t)l * RR * FF;
        const float* Wu = wu + (size_t)l * DDIM * FF;
        const float* Au = au + (size_t)l * DDIM * RR;
        const float* Bu = bu + (size_t)l * RR * FF;
        const float* Wd = wd + (size_t)l * FF * DDIM;
        const float* Ad = ad + (size_t)l * FF * RR;
        const float* Bd = bd + (size_t)l * RR * DDIM;

        // --- attention block ---
        rmsnorm_kernel<<<(unsigned)T, 256, 0, stream>>>(h, ln1 + (size_t)l * DDIM, xn, DDIM);
        lin_lora(stream, xn, Wq, Aq, Bq, t1, q, nullptr, (int)T, DDIM, DDIM);
        lin_lora(stream, xn, Wk, Ak, Bk, t1, k, nullptr, (int)T, DDIM, DDIM);
        lin_lora(stream, xn, Wv, Av, Bv, t1, v, nullptr, (int)T, DDIM, DDIM);
        {
            int n = BB * SS * HH * (DHH / 2);
            rope_kernel<<<(n + 255) / 256, 256, 0, stream>>>(q, k, n);
        }
        // scores[b,h] = (Q[b,:,h,:] x K[b,:,h,:]^T) / sqrt(DH)
        gemm_batched(stream,
                     q, DDIM, (long)SS * DDIM, DHH,
                     k, DDIM, (long)SS * DDIM, DHH, /*transB=*/1,
                     sc, SS, (long)HH * SS * SS, (long)SS * SS,
                     nullptr, SS, 0, 0,
                     SS, SS, DHH, BB * HH, HH, 0.125f);
        softmax_kernel<<<BB * HH * SS, 256, 0, stream>>>(sc, amask);
        // O[b,:,h,:] = attn x V[b,:,h,:]
        gemm_batched(stream,
                     sc, SS, (long)HH * SS * SS, (long)SS * SS,
                     v, DDIM, (long)SS * DDIM, DHH, /*transB=*/0,
                     o, DDIM, (long)SS * DDIM, DHH,
                     nullptr, DDIM, 0, 0,
                     SS, DHH, SS, BB * HH, HH, 1.0f);
        lin_lora(stream, o, Wo, Ao, Bo, t1, h, h, (int)T, DDIM, DDIM);  // h += proj(o)

        // --- SwiGLU MLP block ---
        rmsnorm_kernel<<<(unsigned)T, 256, 0, stream>>>(h, ln2 + (size_t)l * DDIM, xn, DDIM);
        lin_lora(stream, xn, Wg, Ag, Bg, t1, g, nullptr, (int)T, DDIM, FF);
        lin_lora(stream, xn, Wu, Au, Bu, t1, u, nullptr, (int)T, DDIM, FF);
        {
            long n = (long)T * FF;
            silu_mul_kernel<<<(unsigned)((n + 255) / 256), 256, 0, stream>>>(g, u, n);
        }
        lin_lora(stream, g, Wd, Ad, Bd, t1, h, h, (int)T, FF, DDIM);    // h += down(g*u)
    }

    // ---- final norm + heads ----
    rmsnorm_kernel<<<(unsigned)T, 256, 0, stream>>>(h, final_ln, xn, DDIM);
    gemm1(stream, xn, DDIM, lm_head_w, VV, 0, out_logits, VV, nullptr,
          (int)T, VV, DDIM, 1.0f);
    value_head_kernel<<<(unsigned)T, 256, 0, stream>>>(xn, value_w, value_b, out_values);
}
